// DecoderModel_39754217291878
// MI455X (gfx1250) — compile-verified, matrix operand order checked by reading, not run on hardware
//
#include <hip/hip_runtime.h>

// ---------------------------------------------------------------------------
// DecoderModel for MI455X (gfx1250, wave32).
//   B=2, S=2048, D=512, V=32000, H=11
// Dominant cost: final [4096x512]@[512x32000] GEMM (134 GFLOP, 524MB fp32 out)
//   -> bf16 WMMA (v_wmma_f32_16x16x32_bf16), f32 accumulate, pre-packed B,
//      software-pipelined (double-buffered fragments), 32Mx64N wave tile.
// Attention path (H=11) is <1% of FLOPs -> plain VALU kernels.
// ---------------------------------------------------------------------------

#define Bsz 2
#define Sd  2048
#define Dd  512
#define Vd  32000
#define Hd  11
#define ROWS (Bsz*Sd)            // 4096

typedef __bf16        v16bf __attribute__((ext_vector_type(16)));
typedef float         v8f   __attribute__((ext_vector_type(8)));
typedef unsigned int  u32x4 __attribute__((ext_vector_type(4)));

union FragU { v16bf f; u32x4 u[2]; };

// ---- workspace layout (bytes) ---------------------------------------------
#define OFF_BUF0   ((size_t)0)                         // fp32 [4096,512]  8.39MB
#define OFF_BUF1   ((size_t)8388608)                   // fp32 [4096,512]  8.39MB
#define OFF_QKV    ((size_t)16777216)                  // fp32 [4096,33]   0.54MB
#define OFF_ATT    ((size_t)17317888)                  // fp32 [2,2048,2048] 33.5MB (reused as packed-B bf16 32.8MB)
#define OFF_ATTOUT ((size_t)50872320)                  // fp32 [4096,11]   0.18MB
#define OFF_ABF16  ((size_t)51052544)                  // bf16 [4096,512]  4.19MB
// total ~55.2MB

__device__ __forceinline__ unsigned short f32_to_bf16(float f) {
    unsigned int u = __float_as_uint(f);
    unsigned int r = u + 0x7FFFu + ((u >> 16) & 1u);   // round-to-nearest-even
    return (unsigned short)(r >> 16);
}

__device__ __forceinline__ void load_frag(FragU& f, const unsigned short* p) {
    f.u[0] = *(const u32x4*)(p);        // lane K = kbase + 0..7
    f.u[1] = *(const u32x4*)(p + 16);   // lane K = kbase + 16..23 (ISA A layout)
}

__device__ __forceinline__ void load_fragB(FragU& f, const unsigned short* p) {
    // packed-B fragments are 32 contiguous bytes per lane
    f.u[0] = *(const u32x4*)(p);
    f.u[1] = *(const u32x4*)(p + 8);
}

// ---- 1. embedding + sinusoidal positions ----------------------------------
__global__ __launch_bounds__(256) void embed_kernel(const int* __restrict__ x,
                                                    const float* __restrict__ table,
                                                    float* __restrict__ out) {
    int idx = blockIdx.x * 256 + threadIdx.x;          // [0, 4096*512)
    int c = idx & (Dd - 1);
    int s = (idx >> 9) & (Sd - 1);
    int b = idx >> 20;
    int tok = x[b * Sd + s];
    int i = c & ~1;                                    // even index itself
    // denom = 10000^(2i/D) -> inv = exp(-(2i/D)*ln 10000)
    float inv = __expf(-(float)i * (2.0f / (float)Dd) * 9.21034037197618f);
    float ang = (float)s * inv;
    float pv = (c & 1) ? __cosf(ang) : __sinf(ang);
    out[idx] = table[(size_t)tok * Dd + c] + pv;
}

// ---- 2a. fused K/Q/V projection (j: 0-10 K, 11-21 Q, 22-32 V) -------------
__global__ __launch_bounds__(256) void qkv_kernel(const float* __restrict__ in,
                                                  const float* __restrict__ kw, const float* __restrict__ kb,
                                                  const float* __restrict__ qw, const float* __restrict__ qb,
                                                  const float* __restrict__ vw, const float* __restrict__ vb,
                                                  float* __restrict__ qkv) {
    int idx = blockIdx.x * 256 + threadIdx.x;          // [0, 4096*33)
    int row = idx / 33, j = idx - row * 33;
    const float* w; const float* bias; int h;
    if (j < 11)      { w = kw; bias = kb; h = j; }
    else if (j < 22) { w = qw; bias = qb; h = j - 11; }
    else             { w = vw; bias = vb; h = j - 22; }
    const float* e = in + (size_t)row * Dd;
    float acc = bias[h];
    for (int d = 0; d < Dd; ++d) acc += e[d] * w[d * Hd + h];
    qkv[(size_t)row * 33 + j] = acc;
}

// ---- 2b. causal scores (lower triangle only) ------------------------------
__global__ __launch_bounds__(256) void scores_kernel(const float* __restrict__ qkv,
                                                     float* __restrict__ att) {
    int idx = blockIdx.x * 256 + threadIdx.x;          // b*2^22 + q*2^11 + k
    int k = idx & (Sd - 1);
    int q = (idx >> 11) & (Sd - 1);
    int b = idx >> 22;
    if (k > q) return;
    const float* qv = qkv + ((size_t)(b * Sd + q)) * 33 + 11;
    const float* kv = qkv + ((size_t)(b * Sd + k)) * 33;
    float acc = 0.f;
#pragma unroll
    for (int h = 0; h < Hd; ++h) acc += qv[h] * kv[h];
    att[(size_t)idx] = acc;
}

// ---- 2c. softmax over the QUERY axis (columns), masked q>=k ---------------
__global__ __launch_bounds__(256) void colsoftmax_kernel(float* __restrict__ att) {
    __shared__ float red[256];
    int bk = blockIdx.x;                               // b*2048 + k
    int b = bk >> 11, k = bk & (Sd - 1);
    float* base = att + (size_t)b * Sd * Sd + k;       // [b,q,k] at + q*Sd
    int tid = threadIdx.x;
    float m = -3.4e38f;
    for (int q = k + tid; q < Sd; q += 256) m = fmaxf(m, base[(size_t)q * Sd]);
    red[tid] = m; __syncthreads();
    for (int o = 128; o; o >>= 1) { if (tid < o) red[tid] = fmaxf(red[tid], red[tid + o]); __syncthreads(); }
    m = red[0]; __syncthreads();
    float s = 0.f;
    for (int q = k + tid; q < Sd; q += 256) s += __expf(base[(size_t)q * Sd] - m);
    red[tid] = s; __syncthreads();
    for (int o = 128; o; o >>= 1) { if (tid < o) red[tid] += red[tid + o]; __syncthreads(); }
    float inv = 1.0f / red[0];
    for (int q = k + tid; q < Sd; q += 256) {
        size_t a = (size_t)q * Sd;
        base[a] = __expf(base[a] - m) * inv;
    }
}

// ---- 2d. res[b,q,h] = sum_{k<=q} att[b,q,k] * v[b,k,h] --------------------
__global__ __launch_bounds__(256) void attv_kernel(const float* __restrict__ att,
                                                   const float* __restrict__ qkv,
                                                   float* __restrict__ attout) {
    __shared__ float red[256 * Hd];
    int bq = blockIdx.x;                               // b*2048 + q
    int b = bq >> 11, q = bq & (Sd - 1);
    const float* arow = att + (size_t)bq * Sd;
    float acc[Hd];
#pragma unroll
    for (int h = 0; h < Hd; ++h) acc[h] = 0.f;
    for (int k = threadIdx.x; k <= q; k += 256) {
        float a = arow[k];
        const float* v = qkv + ((size_t)(b * Sd + k)) * 33 + 22;
#pragma unroll
        for (int h = 0; h < Hd; ++h) acc[h] += a * v[h];
    }
#pragma unroll
    for (int h = 0; h < Hd; ++h) red[threadIdx.x * Hd + h] = acc[h];
    __syncthreads();
    if (threadIdx.x < Hd) {
        float s = 0.f;
        for (int t = 0; t < 256; ++t) s += red[t * Hd + threadIdx.x];
        attout[(size_t)bq * Hd + threadIdx.x] = s;
    }
}

// ---- 2e. out-projection [4096,11]@[11,512] + bias -------------------------
__global__ __launch_bounds__(256) void outproj_kernel(const float* __restrict__ attout,
                                                      const float* __restrict__ fw,
                                                      const float* __restrict__ fb,
                                                      float* __restrict__ out) {
    int idx = blockIdx.x * 256 + threadIdx.x;          // [0, 4096*512)
    int d = idx & (Dd - 1);
    int row = idx >> 9;
    const float* a = attout + (size_t)row * Hd;
    float acc = fb[d];
#pragma unroll
    for (int h = 0; h < Hd; ++h) acc += a[h] * fw[h * Dd + d];
    out[idx] = acc;
}

// ---- 3a. activations fp32 -> bf16 (row-major, WMMA-A-load friendly) -------
__global__ __launch_bounds__(256) void cvtA_kernel(const float* __restrict__ in,
                                                   unsigned short* __restrict__ out) {
    int idx = blockIdx.x * 256 + threadIdx.x;
    out[idx] = f32_to_bf16(in[idx]);
}

// ---- 3b. pack out_w [512,32000] into per-tile bf16 WMMA-B fragments -------
// tile(nt,kt): 32K x 16N; per-lane 16 bf16 = B[k0..k0+7][n], B[k0+16..k0+23][n]
// with k0 = kt*32 + 8*(lane>>4), n = nt*16 + (lane&15).  1024B per tile.
__global__ __launch_bounds__(256) void packB_kernel(const float* __restrict__ w,
                                                    unsigned short* __restrict__ bp) {
    int idx = blockIdx.x * 256 + threadIdx.x;          // [0, 32000 tiles * 32 lanes)
    int lane = idx & 31;
    int tile = idx >> 5;
    int nt = tile >> 4;                                // tile = nt*16 + kt
    int kt = tile & 15;
    int n = nt * 16 + (lane & 15);
    int koff = (lane >> 4) << 3;
    unsigned short* dst = bp + (size_t)tile * 512 + lane * 16;
#pragma unroll
    for (int e = 0; e < 16; ++e) {
        int v = e >> 1, h = e & 1;
        int k = kt * 32 + koff + ((v < 4) ? ((v << 1) + h) : (16 + ((v - 4) << 1) + h));
        dst[e] = f32_to_bf16(w[(size_t)k * Vd + n]);
    }
}

// ---- 3c. big GEMM: [4096x512]bf16 @ [512x32000]bf16 + bias -> fp32 --------
// 8 waves/block; block tile 64M x 256N; wave tile 32M x 64N; K-step 32.
// Explicit double-buffered fragments: loads for kt+1 issue before the 8 WMMAs
// of kt, so v_wmma never waits on a just-issued global_load_b128.
__global__ __launch_bounds__(256) void gemm_kernel(const unsigned short* __restrict__ A,
                                                   const unsigned short* __restrict__ Bp,
                                                   const float* __restrict__ bias,
                                                   float* __restrict__ out) {
    int tid  = threadIdx.x;
    int wave = tid >> 5;
    int lane = tid & 31;
    int m0   = blockIdx.y * 64 + (wave >> 2) * 32;     // wave's M origin (32 rows)
    int nW   = blockIdx.x * 256 + (wave & 3) * 64;     // wave's N origin (64 cols)
    int ntB  = nW >> 4;                                // first B tile column

    int ksel = (lane >> 4) << 3;
    const unsigned short* aRow0 = A + (size_t)(m0 + (lane & 15)) * Dd + ksel;
    const unsigned short* aRow1 = aRow0 + (size_t)16 * Dd;
    const unsigned short* bBase = Bp + (size_t)ntB * 16 * 512 + lane * 16;

    // C init from bias: C layout -> N = lane&15 (same bias for all 8 M rows)
    v8f c[2][4];
#pragma unroll
    for (int j = 0; j < 4; ++j) {
        float bv = bias[nW + j * 16 + (lane & 15)];
#pragma unroll
        for (int r = 0; r < 8; ++r) { c[0][j][r] = bv; c[1][j][r] = bv; }
    }

    FragU a[2][2], b[2][4];
    load_frag(a[0][0], aRow0);
    load_frag(a[0][1], aRow1);
#pragma unroll
    for (int j = 0; j < 4; ++j) load_fragB(b[0][j], bBase + ((size_t)j * 16) * 512);

#pragma unroll
    for (int kt = 0; kt < 16; ++kt) {
        const int cur = kt & 1, nxt = cur ^ 1;
        if (kt < 15) {
            const unsigned short* ap = aRow0 + (kt + 1) * 32;
            load_frag(a[nxt][0], ap);
            load_frag(a[nxt][1], ap + (size_t)16 * Dd);
#pragma unroll
            for (int j = 0; j < 4; ++j)
                load_fragB(b[nxt][j], bBase + ((size_t)j * 16 + kt + 1) * 512);
        }
        if (kt + 2 < 16) __builtin_prefetch(aRow0 + (kt + 2) * 32, 0, 0); // global_prefetch_b8
#pragma unroll
        for (int j = 0; j < 4; ++j) {
            c[0][j] = __builtin_amdgcn_wmma_f32_16x16x32_bf16(
                          false, a[cur][0].f, false, b[cur][j].f, (short)0, c[0][j], false, false);
            c[1][j] = __builtin_amdgcn_wmma_f32_16x16x32_bf16(
                          false, a[cur][1].f, false, b[cur][j].f, (short)0, c[1][j], false, false);
        }
    }

    // store: C VGPR r -> M = r + 8*(lane>>4), N = lane&15
    int mlo  = m0 + ((lane >> 4) << 3);
    int ncol = lane & 15;
#pragma unroll
    for (int mi = 0; mi < 2; ++mi) {
#pragma unroll
        for (int j = 0; j < 4; ++j) {
            float* o = out + (size_t)(mlo + mi * 16) * Vd + nW + j * 16 + ncol;
#pragma unroll
            for (int r = 0; r < 8; ++r) o[(size_t)r * Vd] = c[mi][j][r];
        }
    }
}

// ---------------------------------------------------------------------------
static void attention_block(const float* in,
                            const float* kw, const float* kb,
                            const float* qw, const float* qb,
                            const float* vw, const float* vb,
                            const float* fw, const float* fb,
                            float* outbuf, float* qkv, float* att, float* attout,
                            hipStream_t stream) {
    qkv_kernel<<<(ROWS * 33) / 256, 256, 0, stream>>>(in, kw, kb, qw, qb, vw, vb, qkv);
    scores_kernel<<<(Bsz * Sd * Sd) / 256, 256, 0, stream>>>(qkv, att);
    colsoftmax_kernel<<<Bsz * Sd, 256, 0, stream>>>(att);
    attv_kernel<<<Bsz * Sd, 256, 0, stream>>>(att, qkv, attout);
    outproj_kernel<<<(ROWS * Dd) / 256, 256, 0, stream>>>(attout, fw, fb, outbuf);
}

extern "C" void kernel_launch(void* const* d_in, const int* in_sizes, int n_in,
                              void* d_out, int out_size, void* d_ws, size_t ws_size,
                              hipStream_t stream) {
    // input order: 0:x 1:emb 2:k1w 3:k1b 4:q1w 5:q1b 6:v1w 7:v1b 8:ff1w 9:ff1b
    //              10:k2w 11:k2b 12:q2w 13:q2b 14:v2w 15:v2b 16:ff2w 17:ff2b
    //              18:out_w 19:out_b
    const int*   x     = (const int*)d_in[0];
    const float* table = (const float*)d_in[1];
    const float* k1w = (const float*)d_in[2];  const float* k1b = (const float*)d_in[3];
    const float* q1w = (const float*)d_in[4];  const float* q1b = (const float*)d_in[5];
    const float* v1w = (const float*)d_in[6];  const float* v1b = (const float*)d_in[7];
    const float* f1w = (const float*)d_in[8];  const float* f1b = (const float*)d_in[9];
    const float* k2w = (const float*)d_in[10]; const float* k2b = (const float*)d_in[11];
    const float* q2w = (const float*)d_in[12]; const float* q2b = (const float*)d_in[13];
    const float* v2w = (const float*)d_in[14]; const float* v2b = (const float*)d_in[15];
    const float* f2w = (const float*)d_in[16]; const float* f2b = (const float*)d_in[17];
    const float* ow  = (const float*)d_in[18]; const float* ob  = (const float*)d_in[19];

    char* ws = (char*)d_ws;
    float* buf0   = (float*)(ws + OFF_BUF0);
    float* buf1   = (float*)(ws + OFF_BUF1);
    float* qkv    = (float*)(ws + OFF_QKV);
    float* att    = (float*)(ws + OFF_ATT);
    float* attout = (float*)(ws + OFF_ATTOUT);
    unsigned short* abf16 = (unsigned short*)(ws + OFF_ABF16);
    unsigned short* bpack = (unsigned short*)(ws + OFF_ATT);   // reuse att region

    // 1. embedding + positions -> buf0
    embed_kernel<<<(ROWS * Dd) / 256, 256, 0, stream>>>(x, table, buf0);
    // 2. attention block 1: buf0 -> buf1
    attention_block(buf0, k1w, k1b, q1w, q1b, v1w, v1b, f1w, f1b,
                    buf1, qkv, att, attout, stream);
    // 3. attention block 2: buf1 -> buf0
    attention_block(buf1, k2w, k2b, q2w, q2b, v2w, v2b, f2w, f2b,
                    buf0, qkv, att, attout, stream);
    // 4. convert activations, pack B (reuses att region -- attention is done)
    cvtA_kernel<<<(ROWS * Dd) / 256, 256, 0, stream>>>(buf0, abf16);
    packB_kernel<<<(2000 * 16 * 32) / 256, 256, 0, stream>>>(ow, bpack);
    // 5. WMMA GEMM -> d_out
    dim3 grid(Vd / 256, ROWS / 64);
    gemm_kernel<<<grid, 256, 0, stream>>>(abf16, bpack, ob, (float*)d_out);
}
